// multiheadAttention_84267258348120
// MI455X (gfx1250) — matrix operand order repaired and checked by reading, not compiled, not run on hardware
//
#include <hip/hip_runtime.h>

// ---------------------------------------------------------------------------
// CDNA5 (gfx1250) fused MHA: bf16 WMMA GEMMs + RoPE epilogue + flash attention
// Async global->LDS staging (ASYNCcnt) and permlane16 softmax reductions.
// ---------------------------------------------------------------------------

#if defined(__has_builtin)
#  if __has_builtin(__builtin_amdgcn_global_load_async_to_lds_b128)
#    define USE_ASYNC_LDS 1
#  endif
#  if __has_builtin(__builtin_amdgcn_s_wait_asynccnt)
#    define HAVE_WAIT_ASYNC_BUILTIN 1
#  endif
#  if __has_builtin(__builtin_amdgcn_permlane16)
#    define USE_PERMLANE16 1
#  endif
#endif

typedef __attribute__((ext_vector_type(16))) __bf16 v16bf;
typedef __attribute__((ext_vector_type(8)))  float  v8f;
typedef __attribute__((ext_vector_type(4)))  int    v4i;

union BFQ  { unsigned long long q[4]; __bf16 h[16]; v16bf v; };
union QW2  { unsigned long long q[2]; __bf16 h[8]; };

__device__ __forceinline__ __bf16 f2bf(float f) {
  union { float f; unsigned u; } a; a.f = f;
  unsigned r = a.u + 0x7FFFu + ((a.u >> 16) & 1u);   // round-to-nearest-even
  union { unsigned short s; __bf16 b; } o;
  o.s = (unsigned short)(r >> 16);
  return o.b;
}

#if defined(USE_ASYNC_LDS)
// 16B per-lane async DMA: global -> LDS, tracked by ASYNCcnt.
typedef __attribute__((address_space(1))) v4i gv4i;   // global 128-bit chunk
typedef __attribute__((address_space(3))) v4i lv4i;   // LDS    128-bit chunk
__device__ __forceinline__ void async_copy16(const void* g, void* l) {
  __builtin_amdgcn_global_load_async_to_lds_b128(
      (gv4i*)(void*)g, (lv4i*)l, 0, 0);
}
__device__ __forceinline__ void wait_async0() {
#if defined(HAVE_WAIT_ASYNC_BUILTIN)
  __builtin_amdgcn_s_wait_asynccnt(0);
#else
  asm volatile("s_wait_asynccnt 0x0" ::: "memory");
#endif
}
#endif

// --- 16-lane-row cross-lane helpers (VALU permlane16, no LDS round trip) ----
#if defined(USE_PERMLANE16)
__device__ __forceinline__ float permf(float v, unsigned s1, unsigned s2) {
  union { float f; unsigned u; } a, r; a.f = v;
  r.u = __builtin_amdgcn_permlane16(a.u, a.u, s1, s2, false, false);
  return r.f;
}
#endif
__device__ __forceinline__ float pair_swap(float v) {   // xor-1 within row
#if defined(USE_PERMLANE16)
  return permf(v, 0x67452301u, 0xEFCDAB89u);
#else
  return __shfl_xor(v, 1, 32);
#endif
}
__device__ __forceinline__ float row_max16(float v) {
#if defined(USE_PERMLANE16)
  v = fmaxf(v, permf(v, 0x67452301u, 0xEFCDAB89u));   // xor 1
  v = fmaxf(v, permf(v, 0x54761032u, 0xDCFE98BAu));   // xor 2
  v = fmaxf(v, permf(v, 0x32107654u, 0xBA98FEDCu));   // xor 4
  v = fmaxf(v, permf(v, 0xFEDCBA98u, 0x76543210u));   // xor 8
#else
  v = fmaxf(v, __shfl_xor(v, 1, 32));
  v = fmaxf(v, __shfl_xor(v, 2, 32));
  v = fmaxf(v, __shfl_xor(v, 4, 32));
  v = fmaxf(v, __shfl_xor(v, 8, 32));
#endif
  return v;
}
__device__ __forceinline__ float row_sum16(float v) {
#if defined(USE_PERMLANE16)
  v += permf(v, 0x67452301u, 0xEFCDAB89u);
  v += permf(v, 0x54761032u, 0xDCFE98BAu);
  v += permf(v, 0x32107654u, 0xBA98FEDCu);
  v += permf(v, 0xFEDCBA98u, 0x76543210u);
#else
  v += __shfl_xor(v, 1, 32);
  v += __shfl_xor(v, 2, 32);
  v += __shfl_xor(v, 4, 32);
  v += __shfl_xor(v, 8, 32);
#endif
  return v;
}

// A-matrix fragment (16x32 bf16): lane half 0 -> K 0..15, half 1 -> K 16..31.
__device__ __forceinline__ v16bf frag_a_load(const __bf16* p) {
  BFQ f;
  const unsigned long long* q = (const unsigned long long*)p;
  f.q[0] = q[0]; f.q[1] = q[1];   // K+0..7
  f.q[2] = q[2]; f.q[3] = q[3];   // K+8..15
  return f.v;
}
// B-matrix fragment (32x16 bf16): 16 contiguous K values from row (lane&15).
__device__ __forceinline__ v16bf frag_b_load(const __bf16* p) {
  BFQ f;
  const unsigned long long* q = (const unsigned long long*)p;
  f.q[0] = q[0]; f.q[1] = q[1]; f.q[2] = q[2]; f.q[3] = q[3];
  return f.v;
}

__device__ __forceinline__ v8f wmma_bf16(v16bf a, v16bf b, v8f c) {
  return __builtin_amdgcn_wmma_f32_16x16x32_bf16(
      false, a, false, b, (short)0, c, false, false);
}

// ---------------------------------------------------------------------------
// One-time f32 -> bf16 conversion (x is read 3x by QKV, weights once each).
// ---------------------------------------------------------------------------
__global__ __launch_bounds__(256)
void cvt_f32_to_bf16(const float* __restrict__ src, __bf16* __restrict__ dst,
                     int n8) {
  int i = blockIdx.x * 256 + threadIdx.x;
  if (i >= n8) return;
  const float4* s = (const float4*)src + (size_t)i * 2;
  float4 a = s[0], b = s[1];
  QW2 u;
  u.h[0] = f2bf(a.x); u.h[1] = f2bf(a.y); u.h[2] = f2bf(a.z); u.h[3] = f2bf(a.w);
  u.h[4] = f2bf(b.x); u.h[5] = f2bf(b.y); u.h[6] = f2bf(b.z); u.h[7] = f2bf(b.w);
  unsigned long long* d = (unsigned long long*)(dst + (size_t)i * 8);
  d[0] = u.q[0]; d[1] = u.q[1];
}

// ---------------------------------------------------------------------------
// GEMM: Out[M x N] = A[M x K] * W[N x K]^T, both bf16 row-major (K contig).
// BM=BN=64, BK=32, 128 threads (2x2 wave grid, 2x2 frags per wave).
// ROPE=1 applies rotary embedding per 64-wide head slice in the epilogue.
// ---------------------------------------------------------------------------
template <int ROPE, int OUT_BF16>
__global__ __launch_bounds__(128)
void gemm_bf16_wmma(const __bf16* __restrict__ A, const __bf16* __restrict__ W,
                    void* __restrict__ Outp, const int* __restrict__ tokpos,
                    int M, int N, int Kd, int S) {
  constexpr int LDT = 40;                 // 32 + 8 pad -> 80B rows, 16B aligned
  __shared__ __bf16 As[64 * LDT];
  __shared__ __bf16 Bs[64 * LDT];
  const int tid   = threadIdx.x;
  const int lane  = tid & 31, wid = tid >> 5;
  const int lhalf = lane >> 4, lcol = lane & 15;
  const int mbase = blockIdx.y * 64, nbase = blockIdx.x * 64;
  const int wm = wid >> 1, wn = wid & 1;
  v8f acc[2][2] = {};

  for (int kk = 0; kk < Kd; kk += 32) {
    __syncthreads();
    // stage A (64x32) and B (64x32) bf16 tiles: 16B chunk per thread per iter
    #pragma unroll
    for (int it = 0; it < 2; ++it) {
      int e   = tid + it * 128;           // 256 chunks of 8 elements
      int row = e >> 2;
      int c8  = (e & 3) * 8;
      const __bf16* ag = A + (size_t)(mbase + row) * Kd + kk + c8;
      const __bf16* bg = W + (size_t)(nbase + row) * Kd + kk + c8;
#if defined(USE_ASYNC_LDS)
      async_copy16(ag, As + row * LDT + c8);
      async_copy16(bg, Bs + row * LDT + c8);
#else
      const unsigned long long* as = (const unsigned long long*)ag;
      unsigned long long* ad = (unsigned long long*)(As + row * LDT + c8);
      ad[0] = as[0]; ad[1] = as[1];
      const unsigned long long* bs = (const unsigned long long*)bg;
      unsigned long long* bd = (unsigned long long*)(Bs + row * LDT + c8);
      bd[0] = bs[0]; bd[1] = bs[1];
#endif
    }
    if (kk + 32 < Kd) {                    // global_prefetch_b8 of next slab
      __builtin_prefetch(A + (size_t)(mbase + (tid >> 1)) * Kd + kk + 32, 0, 1);
      __builtin_prefetch(W + (size_t)(nbase + (tid >> 1)) * Kd + kk + 32, 0, 1);
    }
#if defined(USE_ASYNC_LDS)
    wait_async0();
#endif
    __syncthreads();

    const __bf16* ar = As + (wm * 32 + lcol) * LDT + lhalf * 16;
    v16bf a0 = frag_a_load(ar);
    v16bf a1 = frag_a_load(ar + 16 * LDT);
    const __bf16* br = Bs + (wn * 32 + lcol) * LDT + lhalf * 16;
    v16bf b0 = frag_b_load(br);
    v16bf b1 = frag_b_load(br + 16 * LDT);
    acc[0][0] = wmma_bf16(a0, b0, acc[0][0]);
    acc[0][1] = wmma_bf16(a0, b1, acc[0][1]);
    acc[1][0] = wmma_bf16(a1, b0, acc[1][0]);
    acc[1][1] = wmma_bf16(a1, b1, acc[1][1]);
  }

  // epilogue: C-frag row = e + lhalf*8, col = lcol (per 16x16 tile)
  #pragma unroll
  for (int tm = 0; tm < 2; ++tm)
    #pragma unroll
    for (int tn = 0; tn < 2; ++tn)
      #pragma unroll
      for (int e = 0; e < 8; ++e) {
        int row = mbase + wm * 32 + tm * 16 + e + lhalf * 8;
        int col = nbase + wn * 32 + tn * 16 + lcol;
        float val = acc[tm][tn][e];
        if (ROPE) {
          float other = pair_swap(val);         // rotary partner (col ^ 1)
          float pos = (float)tokpos[row % S];
          int   i   = (col & 63) >> 1;          // pair index within head (64)
          float ang = pos * __powf(10000.0f, (float)(-2 * i) * (1.0f / 64.0f));
          float sn, cs; __sincosf(ang, &sn, &cs);
          val = (col & 1) ? (other * sn + val * cs) : (val * cs - other * sn);
        }
        if (OUT_BF16) ((__bf16*)Outp)[(size_t)row * N + col] = f2bf(val);
        else          ((float*)Outp)[(size_t)row * N + col] = val;
      }
}

// ---------------------------------------------------------------------------
// Flash attention: one workgroup = one (batch, head, 64-query tile).
// 4 waves; each wave owns 16 query rows with resident Q fragments.
// K tiles staged via async DMA; V transposed into LDS; online softmax with
// permlane16 row reductions; P staged through wave-private LDS.
// ---------------------------------------------------------------------------
__global__ __launch_bounds__(128)
void flash_attn_kernel(const __bf16* __restrict__ Q, const __bf16* __restrict__ K,
                       const __bf16* __restrict__ V, __bf16* __restrict__ O,
                       int S, int D) {
  __shared__ __bf16 Ks[32 * 72];       // [key][d]  (144B rows, 16B aligned)
  __shared__ __bf16 Vts[64 * 36];      // [d][key]  (transposed)
  __shared__ __bf16 Ps[4 * 16 * 36];   // per-wave P staging (C -> A layout)

  const int tid   = threadIdx.x;
  const int lane  = tid & 31, w = tid >> 5;
  const int lhalf = lane >> 4, lcol = lane & 15;
  const int b = blockIdx.z, h = blockIdx.y, qt = blockIdx.x;
  const int rb = b * S;
  const int ch = h * 64;
  const int q0 = qt * 64 + w * 16;

  // resident Q fragments for this wave's 16 rows (d 0..31 and 32..63)
  const __bf16* qrow = Q + (size_t)(rb + q0 + lcol) * D + ch + lhalf * 16;
  v16bf aq0 = frag_a_load(qrow);
  v16bf aq1 = frag_a_load(qrow + 32);

  float m[8], l[8];
  v8f o[4] = {};
  #pragma unroll
  for (int e = 0; e < 8; ++e) { m[e] = -3.0e38f; l[e] = 0.0f; }

  const float scale = 0.125f;                 // 1/sqrt(64)
  const int nkt = 2 * (qt + 1);               // causal: keys <= qt*64+63
  for (int t = 0; t < nkt; ++t) {
    const int kb = t * 32;
    __syncthreads();
    // stage K tile (async DMA) + load V chunks; then scatter V transposed
    QW2 vu[2];
    #pragma unroll
    for (int it = 0; it < 2; ++it) {
      int e   = tid + it * 128;               // 256 chunks of 8 over 32x64
      int row = e >> 3;
      int c8  = (e & 7) * 8;
      const __bf16* kg = K + (size_t)(rb + kb + row) * D + ch + c8;
#if defined(USE_ASYNC_LDS)
      async_copy16(kg, Ks + row * 72 + c8);
#else
      const unsigned long long* ks = (const unsigned long long*)kg;
      unsigned long long* kd = (unsigned long long*)(Ks + row * 72 + c8);
      kd[0] = ks[0]; kd[1] = ks[1];
#endif
      const unsigned long long* vs = (const unsigned long long*)
          (V + (size_t)(rb + kb + row) * D + ch + c8);
      vu[it].q[0] = vs[0]; vu[it].q[1] = vs[1];
    }
    #pragma unroll
    for (int it = 0; it < 2; ++it) {
      int e   = tid + it * 128;
      int row = e >> 3;
      int c8  = (e & 7) * 8;
      #pragma unroll
      for (int j = 0; j < 8; ++j) Vts[(c8 + j) * 36 + row] = vu[it].h[j];
    }
    if (t + 1 < nkt) {                        // prefetch next key/value tile
      __builtin_prefetch(K + (size_t)(rb + kb + 32 + (tid >> 2)) * D + ch, 0, 1);
      __builtin_prefetch(V + (size_t)(rb + kb + 32 + (tid >> 2)) * D + ch, 0, 1);
    }
#if defined(USE_ASYNC_LDS)
    wait_async0();
#endif
    __syncthreads();

    // S = Q(16x64) x K^T(64x32): two 16-col halves, K-depth 64 = 2 WMMAs each
    const __bf16* kr0 = Ks + lcol * 72 + lhalf * 16;
    const __bf16* kr1 = kr0 + 16 * 72;
    v8f s0 = {}, s1 = {};
    s0 = wmma_bf16(aq0, frag_b_load(kr0),      s0);
    s0 = wmma_bf16(aq1, frag_b_load(kr0 + 32), s0);
    s1 = wmma_bf16(aq0, frag_b_load(kr1),      s1);
    s1 = wmma_bf16(aq1, frag_b_load(kr1 + 32), s1);

    // online softmax; rows of a C-frag half live in one 16-lane group
    __bf16* pw = Ps + w * (16 * 36);
    #pragma unroll
    for (int e = 0; e < 8; ++e) {
      const int qr = q0 + e + lhalf * 8;
      float v0 = (kb + lcol      <= qr) ? s0[e] * scale : -3.0e38f;
      float v1 = (kb + 16 + lcol <= qr) ? s1[e] * scale : -3.0e38f;
      float mn  = fmaxf(m[e], row_max16(fmaxf(v0, v1)));
      float cor = __expf(m[e] - mn);
      float p0  = __expf(v0 - mn);
      float p1  = __expf(v1 - mn);
      l[e] = l[e] * cor + row_sum16(p0 + p1);
      m[e] = mn;
      #pragma unroll
      for (int tn = 0; tn < 4; ++tn) o[tn][e] *= cor;
      const int pr = (e + lhalf * 8) * 36;
      pw[pr + lcol]      = f2bf(p0);
      pw[pr + 16 + lcol] = f2bf(p1);
    }

    // O += P(16x32) x V(32x64): A from staged P, B from transposed V tile
    v16bf ap = frag_a_load(pw + lcol * 36 + lhalf * 16);
    #pragma unroll
    for (int tn = 0; tn < 4; ++tn) {
      const __bf16* vb = Vts + (tn * 16 + lcol) * 36 + lhalf * 16;
      o[tn] = wmma_bf16(ap, frag_b_load(vb), o[tn]);
    }
  }

  #pragma unroll
  for (int tn = 0; tn < 4; ++tn)
    #pragma unroll
    for (int e = 0; e < 8; ++e) {
      int qr = q0 + e + lhalf * 8;
      int dc = tn * 16 + lcol;
      O[(size_t)(rb + qr) * D + ch + dc] = f2bf(o[tn][e] / l[e]);
    }
}

// ---------------------------------------------------------------------------
extern "C" void kernel_launch(void* const* d_in, const int* in_sizes, int n_in,
                              void* d_out, int out_size, void* d_ws, size_t ws_size,
                              hipStream_t stream) {
  const float* x  = (const float*)d_in[0];
  const int* tpos = (const int*)d_in[1];
  const float* Wq = (const float*)d_in[2];
  const float* Wk = (const float*)d_in[3];
  const float* Wv = (const float*)d_in[4];
  const float* Wo = (const float*)d_in[5];
  const int B = 2, S = 2048, D = 1024, H = 16;
  const int M = B * S;

  size_t actB = (size_t)M * D * sizeof(__bf16);     // 8 MB
  size_t wB   = (size_t)D * D * sizeof(__bf16);     // 2 MB
  char* p = (char*)d_ws;
  __bf16* Qb  = (__bf16*)p; p += actB;
  __bf16* Kb  = (__bf16*)p; p += actB;
  __bf16* Vb  = (__bf16*)p; p += actB;
  __bf16* Ob  = (__bf16*)p; p += actB;
  __bf16* Xb  = (__bf16*)p; p += actB;
  __bf16* Wqb = (__bf16*)p; p += wB;
  __bf16* Wkb = (__bf16*)p; p += wB;
  __bf16* Wvb = (__bf16*)p; p += wB;
  __bf16* Wob = (__bf16*)p; p += wB;

  // one-time bf16 conversion of activations + weights
  int n8a = (M * D) / 8, n8w = (D * D) / 8;
  cvt_f32_to_bf16<<<(n8a + 255) / 256, 256, 0, stream>>>(x,  Xb,  n8a);
  cvt_f32_to_bf16<<<(n8w + 255) / 256, 256, 0, stream>>>(Wq, Wqb, n8w);
  cvt_f32_to_bf16<<<(n8w + 255) / 256, 256, 0, stream>>>(Wk, Wkb, n8w);
  cvt_f32_to_bf16<<<(n8w + 255) / 256, 256, 0, stream>>>(Wv, Wvb, n8w);
  cvt_f32_to_bf16<<<(n8w + 255) / 256, 256, 0, stream>>>(Wo, Wob, n8w);

  dim3 blk(128);
  dim3 gg(D / 64, M / 64);
  gemm_bf16_wmma<1, 1><<<gg, blk, 0, stream>>>(Xb, Wqb, Qb, tpos, M, D, D, S);
  gemm_bf16_wmma<1, 1><<<gg, blk, 0, stream>>>(Xb, Wkb, Kb, tpos, M, D, D, S);
  gemm_bf16_wmma<0, 1><<<gg, blk, 0, stream>>>(Xb, Wvb, Vb, tpos, M, D, D, S);

  dim3 ga(S / 64, H, B);
  flash_attn_kernel<<<ga, blk, 0, stream>>>(Qb, Kb, Vb, Ob, S, D);

  gemm_bf16_wmma<0, 0><<<gg, blk, 0, stream>>>(Ob, Wob, d_out, tpos, M, D, D, S);

  (void)in_sizes; (void)n_in; (void)out_size; (void)ws_size;
}
